// MultiheadAttention_25555055411600
// MI455X (gfx1250) — matrix operand order repaired and checked by reading, not compiled, not run on hardware
//
#include <hip/hip_runtime.h>

// ---------------------------------------------------------------------------
// Types for CDNA5 WMMA
// ---------------------------------------------------------------------------
typedef __attribute__((ext_vector_type(16))) __bf16 v16bf;
typedef __attribute__((ext_vector_type(8)))  float  v8f;
typedef __attribute__((ext_vector_type(4)))  int    v4i;

union FragA { unsigned u[8]; v16bf v; };

__device__ __forceinline__ unsigned short f2bf(float f) {
  unsigned u = __float_as_uint(f);
  unsigned r = u + 0x7fffu + ((u >> 16) & 1u);   // round-to-nearest-even
  return (unsigned short)(r >> 16);
}

// A-fragment (16x32 bf16) per-lane K base for VGPR p, lane-half h
__device__ __forceinline__ int kbaseA(int p, int half) {
  return 2 * (p & 3) + 16 * (p >> 2) + 8 * half;
}
// B-fragment (32x16 bf16) K base (B staged transposed in LDS: [n][k])
__device__ __forceinline__ int kbaseB(int p, int half) {
  return 2 * (p & 3) + 16 * (p >> 2) + 8 * half;
}

// ---------------------------------------------------------------------------
// CDNA5 async global -> LDS copy (ASYNCcnt path); 16B per lane per instr.
// ---------------------------------------------------------------------------
__device__ __forceinline__ void async_g2l_b128(const void* gptr, void* lptr) {
#if __has_builtin(__builtin_amdgcn_global_load_async_to_lds_b128)
  __builtin_amdgcn_global_load_async_to_lds_b128((v4i*)gptr, (v4i*)lptr, 0, 0);
#else
  unsigned           laddr = (unsigned)(unsigned long long)lptr;   // LDS offset
  unsigned long long ga    = (unsigned long long)gptr;
  asm volatile("global_load_async_to_lds_b128 %0, %1, off"
               :: "v"(laddr), "v"(ga) : "memory");
#endif
}

__device__ __forceinline__ void wait_async0() {
#if __has_builtin(__builtin_amdgcn_s_wait_asynccnt)
  __builtin_amdgcn_s_wait_asynccnt(0);
#else
  asm volatile("s_wait_asynccnt 0x0" ::: "memory");
#endif
}

#define T_   1536
#define H_   8
#define K_   64
#define V_   192
#define RKPAD 3072   // 2T-1 = 3071 padded to 3072 (row 3071 = zeros)

// ---------------------------------------------------------------------------
// f32 -> bf16 conversion
// ---------------------------------------------------------------------------
__global__ void cvt_bf16(const float* __restrict__ in,
                         unsigned short* __restrict__ out, int n) {
  int i = blockIdx.x * blockDim.x + threadIdx.x;
  int stride = gridDim.x * blockDim.x;
  for (; i < n; i += stride) out[i] = f2bf(in[i]);
}

// ---------------------------------------------------------------------------
// Positional features (Enformer basis)
// ---------------------------------------------------------------------------
__global__ void gamma_max_kernel(unsigned* __restrict__ gmax) {
  int idx = blockIdx.x * blockDim.x + threadIdx.x;
  if (idx >= 3071 * 32) return;
  int r = idx >> 5, i = idx & 31;
  float ap = fabsf((float)(r - (T_ - 1)));
  float mean = 48.f * (float)(i + 1);
  float conc = (mean / 24.f) * (mean / 24.f);
  float rate = mean / 576.f;
  float prob;
  if (ap <= 0.f) prob = 1e-8f;
  else {
    float lp = (conc - 1.f) * logf(ap) - rate * ap - (lgammaf(conc) - conc * logf(rate));
    prob = expf(lp) + 1e-8f;
  }
  atomicMax(gmax, __float_as_uint(prob));   // prob > 0 -> bit pattern monotone
}

__global__ void pos_emb_kernel(unsigned short* __restrict__ emb,
                               const unsigned* __restrict__ gmaxp) {
  int idx = blockIdx.x * blockDim.x + threadIdx.x;
  if (idx >= RKPAD * 192) return;
  int r = idx / 192, f = idx % 192;
  float val = 0.f;
  if (r < 2 * T_ - 1) {
    float pos = (float)(r - (T_ - 1));
    float ap = fabsf(pos);
    int sym = f % 96, cls = sym / 32, i = sym % 32;
    if (cls == 0) {                       // exponential half-life
      float maxr = __log2f((float)T_);
      float e = 3.f + (maxr - 3.f) * (float)i / 31.f;
      float hl = exp2f(e);
      val = exp2f(-ap / hl);
    } else if (cls == 1) {                // central mask
      float w = exp2f((float)(i + 1)) - 1.f;
      val = (w > ap) ? 1.f : 0.f;
    } else {                              // gamma pdf, normalized by global max
      float gmax = __uint_as_float(*gmaxp);
      float mean = 48.f * (float)(i + 1);
      float conc = (mean / 24.f) * (mean / 24.f);
      float rate = mean / 576.f;
      float prob;
      if (ap <= 0.f) prob = 1e-8f;
      else prob = expf((conc - 1.f) * logf(ap) - rate * ap -
                       (lgammaf(conc) - conc * logf(rate))) + 1e-8f;
      val = prob / gmax;
    }
    if (f >= 96) {
      float sg = (pos > 0.f) ? 1.f : ((pos < 0.f) ? -1.f : 0.f);
      val *= sg;
    }
  }
  emb[idx] = f2bf(val);
}

// ---------------------------------------------------------------------------
// Generic bf16 WMMA GEMM: C[M,N] = A[M,Kd] * B[Kd,N]; A,B row-major bf16.
// Macro tile 128x64, 8 waves (4x2), each wave 32x32 (2x2 WMMA tiles), k-step 32.
// Double-buffered: iteration k issues async A-copy + B loads for k+1, then
// runs WMMAs on the current buffer; one barrier per iteration.
// Epilogue modes: 0 f32 | 1 k->[B,H,T,64] | 2 v->[B,H,T,192] | 3 qw/qr | 5 r_k
// ---------------------------------------------------------------------------
__global__ __launch_bounds__(256)
void gemm_bf16(const unsigned short* __restrict__ A,
               const unsigned short* __restrict__ Bm,
               int M, int N, int Kd, int mode,
               float* __restrict__ Cf,
               unsigned short* __restrict__ C16,
               unsigned short* __restrict__ C16b,
               const float* __restrict__ bw, const float* __restrict__ br,
               int Tdim) {
  __shared__ unsigned short As[2][128][40];  // 80B rows: 16B aligned, 20-bank stride
  __shared__ unsigned short Bs[2][64][34];
  const int tid  = threadIdx.x;
  const int lane = tid & 31;
  const int wave = tid >> 5;
  const int half = lane >> 4, ln = lane & 15;
  const int m0 = blockIdx.x * 128, n0 = blockIdx.y * 64;
  const int wm = wave & 3, wn = wave >> 2;

  // ---- prologue: stage tile 0 into buffer 0
#pragma unroll
  for (int i = 0; i < 2; ++i) {
    int li = tid + i * 256;
    int mr = li >> 2, sg = li & 3;
    async_g2l_b128(A + (size_t)(m0 + mr) * Kd + sg * 8, &As[0][mr][sg * 8]);
  }
  {
    unsigned tb[4];
#pragma unroll
    for (int i = 0; i < 4; ++i) {
      int li = tid + i * 256;
      int kk = li >> 5, c2 = li & 31;
      tb[i] = *(const unsigned*)(Bm + (size_t)kk * N + n0 + 2 * c2);
    }
#pragma unroll
    for (int i = 0; i < 4; ++i) {
      int li = tid + i * 256;
      int kk = li >> 5, c2 = li & 31;
      Bs[0][2 * c2][kk]     = (unsigned short)(tb[i] & 0xffffu);
      Bs[0][2 * c2 + 1][kk] = (unsigned short)(tb[i] >> 16);
    }
  }
  wait_async0();
  __syncthreads();

  v8f acc[2][2] = {};
  for (int kt = 0; kt < Kd; kt += 32) {
    const int cur = (kt >> 5) & 1, nxt = cur ^ 1;
    const bool more = (kt + 32) < Kd;
    unsigned tb[4] = {0u, 0u, 0u, 0u};
    if (more) {
      // async-stage NEXT A tile (safe: buffer nxt was fully consumed before
      // the barrier that ended the previous iteration)
#pragma unroll
      for (int i = 0; i < 2; ++i) {
        int li = tid + i * 256;
        int mr = li >> 2, sg = li & 3;
        async_g2l_b128(A + (size_t)(m0 + mr) * Kd + kt + 32 + sg * 8,
                       &As[nxt][mr][sg * 8]);
      }
      // batch-load NEXT B tile into regs
#pragma unroll
      for (int i = 0; i < 4; ++i) {
        int li = tid + i * 256;
        int kk = li >> 5, c2 = li & 31;
        tb[i] = *(const unsigned*)(Bm + (size_t)(kt + 32 + kk) * N + n0 + 2 * c2);
      }
      if (kt + 64 < Kd) {   // global_prefetch_b8 two tiles ahead
        __builtin_prefetch(A + (size_t)(m0 + (tid >> 1)) * Kd + kt + 64, 0, 0);
        __builtin_prefetch(Bm + (size_t)(kt + 64 + (tid >> 3)) * N + n0, 0, 0);
      }
    }

    // ---- compute on current buffer
    FragA af[2], bfr[2];
#pragma unroll
    for (int s = 0; s < 2; ++s) {
      int mrow = wm * 32 + s * 16 + ln;
#pragma unroll
      for (int p = 0; p < 8; ++p)
        af[s].u[p] = *(const unsigned*)&As[cur][mrow][kbaseA(p, half)];
      int ncol = wn * 32 + s * 16 + ln;
#pragma unroll
      for (int p = 0; p < 8; ++p)
        bfr[s].u[p] = *(const unsigned*)&Bs[cur][ncol][kbaseB(p, half)];
    }
#pragma unroll
    for (int sm = 0; sm < 2; ++sm)
#pragma unroll
      for (int sn = 0; sn < 2; ++sn)
        acc[sm][sn] = __builtin_amdgcn_wmma_f32_16x16x32_bf16(
            false, af[sm].v, false, bfr[sn].v, (short)0, acc[sm][sn], false, false);

    if (more) {
      // transpose-store NEXT B tile, fence this wave's async copies
#pragma unroll
      for (int i = 0; i < 4; ++i) {
        int li = tid + i * 256;
        int kk = li >> 5, c2 = li & 31;
        Bs[nxt][2 * c2][kk]     = (unsigned short)(tb[i] & 0xffffu);
        Bs[nxt][2 * c2 + 1][kk] = (unsigned short)(tb[i] >> 16);
      }
      wait_async0();
    }
    __syncthreads();
  }

  const float scale = 0.125f;  // K^-0.5
#pragma unroll
  for (int sm = 0; sm < 2; ++sm)
#pragma unroll
    for (int sn = 0; sn < 2; ++sn)
#pragma unroll
      for (int vv = 0; vv < 8; ++vv) {
        int R  = m0 + wm * 32 + sm * 16 + vv + 8 * half;
        int Cc = n0 + wn * 32 + sn * 16 + ln;
        float val = acc[sm][sn][vv];
        if (mode == 0) {
          Cf[(size_t)R * N + Cc] = val;
        } else if (mode == 1) {
          int b = R / Tdim, t = R % Tdim, h = Cc >> 6, kk = Cc & 63;
          C16[(((size_t)(b * H_ + h) * Tdim) + t) * 64 + kk] = f2bf(val);
        } else if (mode == 2) {
          int b = R / Tdim, t = R % Tdim, h = Cc / 192, vc = Cc % 192;
          C16[(((size_t)(b * H_ + h) * Tdim) + t) * 192 + vc] = f2bf(val);
        } else if (mode == 3) {
          int b = R / Tdim, t = R % Tdim, h = Cc >> 6, kk = Cc & 63;
          size_t o = (((size_t)(b * H_ + h) * Tdim) + t) * 64 + kk;
          C16[o]  = f2bf(val * scale + bw[h * 64 + kk]);
          C16b[o] = f2bf(val * scale + br[h * 64 + kk]);
        } else {  // mode 5
          int h = Cc >> 6, kk = Cc & 63;
          C16[((size_t)h * RKPAD + R) * 64 + kk] = f2bf(val);
        }
      }
}

// ---------------------------------------------------------------------------
// Fused flash attention with relative-position Toeplitz band.
// Grid: x = T/128 query blocks, y = 2 (V halves of 96), z = B*H.
// 8 waves; wave w owns 16 query rows. Key blocks of 32.
// logits[i,j] = qw_i . k_j  +  qr_i . r_k[j - i + T - 1]
// K and r_k tiles staged with async global->LDS; V via batched reg transpose.
// ---------------------------------------------------------------------------
__global__ __launch_bounds__(256)
void attn_kernel(const unsigned short* __restrict__ qw,
                 const unsigned short* __restrict__ qr,
                 const unsigned short* __restrict__ kb,
                 const unsigned short* __restrict__ vb,
                 const unsigned short* __restrict__ rk,
                 unsigned short* __restrict__ outb) {
  __shared__ unsigned short Ks[32][72];    // [key n][kk]; 144B rows (16B aligned)
  __shared__ unsigned short RKs[160][72];  // union r_k band for all 8 waves
  __shared__ unsigned short Vs[96][34];    // [v col][key]  (= B^T layout)
  __shared__ float WS[8][16][48];          // per-wave band tile; aliased as P

  const int tid  = threadIdx.x;
  const int lane = tid & 31, wave = tid >> 5;
  const int half = lane >> 4, ln = lane & 15;
  const int bh = blockIdx.z, h = bh & 7, b = bh >> 3;
  const int iblk = blockIdx.x * 128;
  const int iw = iblk + wave * 16;        // wave's first query row
  const int vbase = blockIdx.y * 96;

  const unsigned short* qwp = qw + (size_t)bh * T_ * 64;
  const unsigned short* qrp = qr + (size_t)bh * T_ * 64;
  const unsigned short* kp  = kb + (size_t)bh * T_ * 64;
  const unsigned short* vp  = vb + (size_t)bh * T_ * 192;
  const unsigned short* rkp = rk + (size_t)h * RKPAD * 64;

  // preload Q fragments (K=64 -> 2 k-steps), live across the whole key loop
  FragA aQw[2], aQr[2];
#pragma unroll
  for (int ks = 0; ks < 2; ++ks)
#pragma unroll
    for (int p = 0; p < 8; ++p) {
      int kk = ks * 32 + kbaseA(p, half);
      aQw[ks].u[p] = *(const unsigned*)(qwp + (size_t)(iw + ln) * 64 + kk);
      aQr[ks].u[p] = *(const unsigned*)(qrp + (size_t)(iw + ln) * 64 + kk);
    }

  v8f out[6] = {};
  float mrow[8], lrow[8];
#pragma unroll
  for (int vv = 0; vv < 8; ++vv) { mrow[vv] = -1e30f; lrow[vv] = 0.f; }

  for (int j0 = 0; j0 < T_; j0 += 32) {
    // ---- async-copy K block: 32 rows x 128B = 256 segs, 1 per thread
    {
      int row = tid >> 3, sg = tid & 7;
      async_g2l_b128(kp + (size_t)(j0 + row) * 64 + sg * 8, &Ks[row][sg * 8]);
    }
    // ---- async-copy union r_k band: 160 rows x 128B = 1280 segs, 5 per thread
    int rall = j0 - iblk + T_ - 128;
#pragma unroll
    for (int i = 0; i < 5; ++i) {
      int li = tid + i * 256;
      int rr = li >> 3, sg = li & 7;
      async_g2l_b128(rkp + (size_t)(rall + rr) * 64 + sg * 8, &RKs[rr][sg * 8]);
    }
    // ---- V block: batch global loads into regs, transpose-store to LDS
    unsigned tv[6];
#pragma unroll
    for (int i = 0; i < 6; ++i) {
      int li = tid + i * 256;
      int kk = li / 48, c2 = li % 48;
      tv[i] = *(const unsigned*)(vp + (size_t)(j0 + kk) * 192 + vbase + 2 * c2);
    }
#pragma unroll
    for (int i = 0; i < 6; ++i) {
      int li = tid + i * 256;
      int kk = li / 48, c2 = li % 48;
      Vs[2 * c2][kk]     = (unsigned short)(tv[i] & 0xffffu);
      Vs[2 * c2 + 1][kk] = (unsigned short)(tv[i] >> 16);
    }
    wait_async0();
    __syncthreads();

    // ---- content tile: (qw) x K^T   [16 x 32]
    v8f cf[2] = {};
#pragma unroll
    for (int n0 = 0; n0 < 2; ++n0) {
      int ncol = n0 * 16 + ln;
#pragma unroll
      for (int ks = 0; ks < 2; ++ks) {
        FragA bk;
#pragma unroll
        for (int p = 0; p < 8; ++p)
          bk.u[p] = *(const unsigned*)&Ks[ncol][ks * 32 + kbaseB(p, half)];
        cf[n0] = __builtin_amdgcn_wmma_f32_16x16x32_bf16(
            false, aQw[ks].v, false, bk.v, (short)0, cf[n0], false, false);
      }
    }
    // ---- band tile: (qr) x r_k^T   [16 x 48], wave-local rows in RKs
    int rowoff = 112 - 16 * wave;
#pragma unroll
    for (int n0 = 0; n0 < 3; ++n0) {
      v8f bnd = {};
      int crow = rowoff + n0 * 16 + ln;
#pragma unroll
      for (int ks = 0; ks < 2; ++ks) {
        FragA br_;
#pragma unroll
        for (int p = 0; p < 8; ++p)
          br_.u[p] = *(const unsigned*)&RKs[crow][ks * 32 + kbaseB(p, half)];
        bnd = __builtin_amdgcn_wmma_f32_16x16x32_bf16(
            false, aQr[ks].v, false, br_.v, (short)0, bnd, false, false);
      }
#pragma unroll
      for (int vv = 0; vv < 8; ++vv)
        WS[wave][vv + 8 * half][n0 * 16 + ln] = bnd[vv];
    }
    __syncthreads();   // band tiles visible

    // ---- apply relative shift: logits += band[di][jo - di + 15]
#pragma unroll
    for (int n0 = 0; n0 < 2; ++n0)
#pragma unroll
      for (int vv = 0; vv < 8; ++vv) {
        int di = vv + 8 * half;
        int c = n0 * 16 + ln + 15 - di;   // in [0,46]
        cf[n0][vv] += WS[wave][di][c];
      }

    // ---- online softmax (rows live in 16-lane halves)
    float pm[8];
#pragma unroll
    for (int vv = 0; vv < 8; ++vv) {
      float mx = fmaxf(cf[0][vv], cf[1][vv]);
#pragma unroll
      for (int mk = 1; mk <= 8; mk <<= 1)
        mx = fmaxf(mx, __shfl_xor(mx, mk, 32));
      float mnew = fmaxf(mrow[vv], mx);
      float sc = __expf(mrow[vv] - mnew);
      float p0 = __expf(cf[0][vv] - mnew);
      float p1 = __expf(cf[1][vv] - mnew);
      cf[0][vv] = p0; cf[1][vv] = p1;
      float rs = p0 + p1;
#pragma unroll
      for (int mk = 1; mk <= 8; mk <<= 1)
        rs += __shfl_xor(rs, mk, 32);
      lrow[vv] = lrow[vv] * sc + rs;
      mrow[vv] = mnew;
      pm[vv] = sc;
    }
#pragma unroll
    for (int t = 0; t < 6; ++t)
#pragma unroll
      for (int vv = 0; vv < 8; ++vv)
        out[t][vv] *= pm[vv];

    // ---- P (C-layout) -> LDS -> A-fragment (reuse per-wave WS slab)
    unsigned short* ps = (unsigned short*)&WS[wave][0][0];
#pragma unroll
    for (int n0 = 0; n0 < 2; ++n0)
#pragma unroll
      for (int vv = 0; vv < 8; ++vv)
        ps[(vv + 8 * half) * 34 + n0 * 16 + ln] = f2bf(cf[n0][vv]);
    __syncthreads();   // keep waves in lockstep

    FragA aP;
#pragma unroll
    for (int p = 0; p < 8; ++p)
      aP.u[p] = *(const unsigned*)&ps[ln * 34 + kbaseA(p, half)];

    // ---- out += P x V   [16 x 96]
#pragma unroll
    for (int t = 0; t < 6; ++t) {
      FragA bv;
      int vcol = t * 16 + ln;
#pragma unroll
      for (int p = 0; p < 8; ++p)
        bv.u[p] = *(const unsigned*)&Vs[vcol][kbaseB(p, half)];
      out[t] = __builtin_amdgcn_wmma_f32_16x16x32_bf16(
          false, aP.v, false, bv.v, (short)0, out[t], false, false);
    }
    __syncthreads();   // all LDS reads done before restaging
  }

  // ---- epilogue: out[i, v] / l_i, written as [B, T, H*V] bf16
#pragma unroll
  for (int t = 0; t < 6; ++t)
#pragma unroll
    for (int vv = 0; vv < 8; ++vv) {
      int di = vv + 8 * half;
      int row = b * T_ + iw + di;
      int col = h * V_ + vbase + t * 16 + ln;
      outb[(size_t)row * (H_ * V_) + col] = f2bf(out[t][vv] / lrow[vv]);
    }
}

// ---------------------------------------------------------------------------
// Host-side orchestration
// ---------------------------------------------------------------------------
extern "C" void kernel_launch(void* const* d_in, const int* in_sizes, int n_in,
                              void* d_out, int out_size, void* d_ws, size_t ws_size,
                              hipStream_t stream) {
  (void)in_sizes; (void)n_in; (void)out_size; (void)ws_size;
  const float* x    = (const float*)d_in[0];
  const float* Wq   = (const float*)d_in[1];
  const float* Wk   = (const float*)d_in[2];
  const float* Wv   = (const float*)d_in[3];
  const float* Wrk  = (const float*)d_in[4];
  const float* Wemb = (const float*)d_in[5];
  const float* rwb  = (const float*)d_in[6];
  const float* rrb  = (const float*)d_in[7];

  char* ws = (char*)d_ws;
  size_t off = 0;
  auto alloc = [&](size_t bytes) -> void* {
    void* p = ws + off;
    off = (off + bytes + 255) & ~(size_t)255;
    return p;
  };

  const int BT = 2 * T_;   // 3072 rows
  unsigned short* x_bf    = (unsigned short*)alloc((size_t)BT * 1536 * 2);
  unsigned short* Wq_bf   = (unsigned short*)alloc((size_t)1536 * 512 * 2);
  unsigned short* Wk_bf   = (unsigned short*)alloc((size_t)1536 * 512 * 2);
  unsigned short* Wv_bf   = (unsigned short*)alloc((size_t)1536 * 1536 * 2);
  unsigned short* Wrk_bf  = (unsigned short*)alloc((size_t)192 * 512 * 2);
  unsigned short* Wemb_bf = (unsigned short*)alloc((size_t)1536 * 1536 * 2);
  unsigned short* emb_bf  = (unsigned short*)alloc((size_t)RKPAD * 192 * 2);
  unsigned short* qw_bf   = (unsigned short*)alloc((size_t)2 * H_ * T_ * 64 * 2);
  unsigned short* qr_bf   = (unsigned short*)alloc((size_t)2 * H_ * T_ * 64 * 2);
  unsigned short* k_bf    = (unsigned short*)alloc((size_t)2 * H_ * T_ * 64 * 2);
  unsigned short* v_bf    = (unsigned short*)alloc((size_t)2 * H_ * T_ * 192 * 2);
  unsigned short* rk_bf   = (unsigned short*)alloc((size_t)H_ * RKPAD * 64 * 2);
  unsigned short* attn_bf = (unsigned short*)alloc((size_t)BT * 1536 * 2);
  unsigned*       gmax    = (unsigned*)alloc(256);

  auto cvt = [&](const float* src, unsigned short* dst, int n) {
    int blk = (n + 255) / 256; if (blk > 2048) blk = 2048;
    cvt_bf16<<<blk, 256, 0, stream>>>(src, dst, n);
  };
  cvt(x,    x_bf,    BT * 1536);
  cvt(Wq,   Wq_bf,   1536 * 512);
  cvt(Wk,   Wk_bf,   1536 * 512);
  cvt(Wv,   Wv_bf,   1536 * 1536);
  cvt(Wrk,  Wrk_bf,  192 * 512);
  cvt(Wemb, Wemb_bf, 1536 * 1536);

  (void)hipMemsetAsync(gmax, 0, 4, stream);
  gamma_max_kernel<<<(3071 * 32 + 255) / 256, 256, 0, stream>>>(gmax);
  pos_emb_kernel<<<(RKPAD * 192 + 255) / 256, 256, 0, stream>>>(emb_bf, gmax);

  // projections
  gemm_bf16<<<dim3(BT / 128, 512 / 64), 256, 0, stream>>>(
      x_bf, Wq_bf, BT, 512, 1536, 3, nullptr, qw_bf, qr_bf, rwb, rrb, T_);
  gemm_bf16<<<dim3(BT / 128, 512 / 64), 256, 0, stream>>>(
      x_bf, Wk_bf, BT, 512, 1536, 1, nullptr, k_bf, nullptr, nullptr, nullptr, T_);
  gemm_bf16<<<dim3(BT / 128, 1536 / 64), 256, 0, stream>>>(
      x_bf, Wv_bf, BT, 1536, 1536, 2, nullptr, v_bf, nullptr, nullptr, nullptr, T_);
  // r_k = emb @ Wrk  (M = 3072 incl. zero pad row 3071)
  gemm_bf16<<<dim3(RKPAD / 128, 512 / 64), 256, 0, stream>>>(
      emb_bf, Wrk_bf, RKPAD, 512, 192, 5, nullptr, rk_bf, nullptr, nullptr, nullptr, T_);

  // fused flash attention with relative-shift band
  attn_kernel<<<dim3(T_ / 128, 2, 2 * H_), 256, 0, stream>>>(
      qw_bf, qr_bf, k_bf, v_bf, rk_bf, attn_bf);

  // final projection -> f32 output
  gemm_bf16<<<dim3(BT / 128, 1536 / 64), 256, 0, stream>>>(
      attn_bf, Wemb_bf, BT, 1536, 1536, 0, (float*)d_out, nullptr, nullptr,
      nullptr, nullptr, T_);
}
